// Branch_2_36386962932308
// MI455X (gfx1250) — compile-verified, matrix operand order checked by reading, not run on hardware
//
#include <hip/hip_runtime.h>
#include <cstddef>
#include <cstdint>

// ---------------------------------------------------------------------------
// CDNA5 (gfx1250) fused 2-stage Mamba block.
// - 5 GEMMs via v_wmma_f32_16x16x32_bf16, operands stored bf16 in global
//   (weights converted once; activations emitted bf16 by producer epilogues)
// - tiles staged LDS-direct with global_load_async_to_lds_b128 (ASYNCcnt),
//   double-buffered: copy of stage s+1 overlaps WMMA of stage s
// - chunked (64x64) parallel scan for the linear recurrence
// - LDS-tiled transposes, LayerNorm with fused bf16 re-emit
// Workspace peak ~174 MB (cycles through MI455X 192MB L2).
// ---------------------------------------------------------------------------

typedef __attribute__((ext_vector_type(16))) __bf16 v16bf;
typedef __attribute__((ext_vector_type(8)))  float  v8f;

__device__ __forceinline__ unsigned short f2bf(float f) {
  unsigned int u = __builtin_bit_cast(unsigned int, f);
  u += 0x7fffu + ((u >> 16) & 1u);            // round-to-nearest-even
  return (unsigned short)(u >> 16);
}
__device__ __forceinline__ __bf16 us2bf(unsigned short s) {
  return __builtin_bit_cast(__bf16, s);
}
__device__ __forceinline__ float silu_f(float x) { return x / (1.0f + __expf(-x)); }
__device__ __forceinline__ float softplus_f(float x) {
  return (x > 20.0f) ? x : logf(1.0f + __expf(x));
}

// ---------------------------------------------------------------------------
// fp32 -> bf16 bulk convert (weights, once per launch)
// ---------------------------------------------------------------------------
__global__ void f32_to_bf16_kernel(const float* __restrict__ in,
                                   unsigned short* __restrict__ out, int n) {
  int i = blockIdx.x * blockDim.x + threadIdx.x;
  if (i < n) out[i] = f2bf(in[i]);
}

// ---------------------------------------------------------------------------
// WMMA GEMM: out[M,N] = A[M,K] @ W[N,K]^T (+bias) (optional SiLU)
// A, W are bf16 in global. Block: 128 thr = 4 waves; tile 64x64; K-stage 64
// (2 wmma k-slices); double-buffered LDS filled by async-to-LDS copies.
// M%64==0, N%64==0, K%64==0 guaranteed by caller.
// ---------------------------------------------------------------------------
#define TLDS  72            // bf16 elems per LDS row (64 + 8 pad; rows 16B-aligned)
#define TILEH (64 * TLDS)   // halves per 64x64 tile

__global__ __launch_bounds__(128)
void gemm_bf16_wmma(const unsigned short* __restrict__ A,
                    const unsigned short* __restrict__ W,
                    const float* __restrict__ bias,
                    float* __restrict__ outf, unsigned short* __restrict__ outb,
                    int M, int N, int K, int act) {
  __shared__ alignas(16) unsigned short lsA[2][TILEH];
  __shared__ alignas(16) unsigned short lsB[2][TILEH];

  const int tid  = threadIdx.x;
  const int lane = tid & 31;
  const int wv   = tid >> 5;
  const int wm   = wv >> 1, wn = wv & 1;
  const int hi   = lane >> 4;
  const int r16  = lane & 15;
  const int bm   = blockIdx.x * 64;
  const int bn   = blockIdx.y * 64;

  const unsigned baseA[2] = { (unsigned)(uintptr_t)&lsA[0][0],
                              (unsigned)(uintptr_t)&lsA[1][0] };
  const unsigned baseB[2] = { (unsigned)(uintptr_t)&lsB[0][0],
                              (unsigned)(uintptr_t)&lsB[1][0] };

  // Stage one 64x64 bf16 tile: 512 16B chunks, 4 per thread, LDS-direct.
  auto issue = [&](const unsigned short* gbase, int rbase, unsigned ldsbase, int k0) {
    #pragma unroll
    for (int i = 0; i < 4; ++i) {
      const int c   = tid + i * 128;
      const int row = c >> 3;            // 8 chunks per 128B row
      const int cb  = (c & 7) * 8;       // bf16 elem offset (16B chunks)
      const unsigned long long g =
          (unsigned long long)(uintptr_t)(gbase + (size_t)(rbase + row) * K + k0 + cb);
      const unsigned lds = ldsbase + (unsigned)(row * TLDS + cb) * 2u;
      asm volatile("global_load_async_to_lds_b128 %0, %1, off"
                   :: "v"(lds), "v"(g) : "memory");
    }
  };

  v8f acc[2][2];
  acc[0][0] = 0.0f; acc[0][1] = 0.0f; acc[1][0] = 0.0f; acc[1][1] = 0.0f;

  const int nk = K >> 6;                 // K / 64 stages
  issue(A, bm, baseA[0], 0);
  issue(W, bn, baseB[0], 0);

  for (int s = 0; s < nk; ++s) {
    asm volatile("s_wait_asynccnt 0x0" ::: "memory");  // my copies for stage s done
    __syncthreads();                                   // everyone's copies done
    if (s + 1 < nk) {                                  // overlap: stage s+1 copy
      issue(A, bm, baseA[(s + 1) & 1], (s + 1) << 6);
      issue(W, bn, baseB[(s + 1) & 1], (s + 1) << 6);
    }
    const int sb = s & 1;

    // Two wmma k-slices (K=32 each) from this 64-wide stage.
    #pragma unroll
    for (int ks = 0; ks < 2; ++ks) {
      v16bf afr[2], bfr[2];
      #pragma unroll
      for (int t = 0; t < 2; ++t) {
        const int ra = (wm * 32 + t * 16 + r16) * TLDS + ks * 32;
        const int rb = (wn * 32 + t * 16 + r16) * TLDS + ks * 32;
        #pragma unroll
        for (int vv = 0; vv < 8; ++vv) {
          // CDNA5 16-bit A/B fragment: elem j=2v+p at k=(v>>2)*16 + hi*8 + (v&3)*2 + p
          const int kb = ((vv >> 2) << 4) + (hi << 3) + ((vv & 3) << 1);
          const unsigned ua = *reinterpret_cast<const unsigned int*>(&lsA[sb][ra + kb]);
          const unsigned ub = *reinterpret_cast<const unsigned int*>(&lsB[sb][rb + kb]);
          afr[t][2 * vv + 0] = us2bf((unsigned short)(ua & 0xffffu));
          afr[t][2 * vv + 1] = us2bf((unsigned short)(ua >> 16));
          bfr[t][2 * vv + 0] = us2bf((unsigned short)(ub & 0xffffu));
          bfr[t][2 * vv + 1] = us2bf((unsigned short)(ub >> 16));
        }
      }
      #pragma unroll
      for (int tm = 0; tm < 2; ++tm)
        #pragma unroll
        for (int tn = 0; tn < 2; ++tn)
          acc[tm][tn] = __builtin_amdgcn_wmma_f32_16x16x32_bf16(
              false, afr[tm], false, bfr[tn], (short)0, acc[tm][tn],
              false, false);
    }
  }

  // Epilogue: D layout -> lane holds n = lane&15, VGPR v holds m = v + 8*hi.
  #pragma unroll
  for (int tm = 0; tm < 2; ++tm) {
    #pragma unroll
    for (int tn = 0; tn < 2; ++tn) {
      const int gn = bn + wn * 32 + tn * 16 + r16;
      #pragma unroll
      for (int vv = 0; vv < 8; ++vv) {
        const int gm = bm + wm * 32 + tm * 16 + 8 * hi + vv;
        float d = acc[tm][tn][vv];
        if (bias) d += bias[gn];
        if (act) d = silu_f(d);
        if (outf) outf[(size_t)gm * N + gn] = d;
        if (outb) outb[(size_t)gm * N + gn] = f2bf(d);
      }
    }
  }
}

// ---------------------------------------------------------------------------
// NCHW(8,C,4096) -> (8,4096,C) bf16 (feeds GEMM only), and fp32 back.
// ---------------------------------------------------------------------------
__global__ void transpose_in_kernel(const float* __restrict__ x,
                                    unsigned short* __restrict__ t,
                                    int C, int HW) {
  __shared__ float tile[32][33];
  const int b = blockIdx.z, l0 = blockIdx.x * 32, c0 = blockIdx.y * 32;
  const int tx = threadIdx.x, ty = threadIdx.y;
  #pragma unroll
  for (int j = 0; j < 32; j += 8)
    tile[ty + j][tx] = x[((size_t)b * C + (c0 + ty + j)) * HW + (l0 + tx)];
  __syncthreads();
  #pragma unroll
  for (int j = 0; j < 32; j += 8)
    t[((size_t)b * HW + (l0 + ty + j)) * C + (c0 + tx)] = f2bf(tile[tx][ty + j]);
}

__global__ void transpose_out_kernel(const float* __restrict__ t, float* __restrict__ o,
                                     int C, int HW) {
  __shared__ float tile[32][33];
  const int b = blockIdx.z, l0 = blockIdx.x * 32, c0 = blockIdx.y * 32;
  const int tx = threadIdx.x, ty = threadIdx.y;
  #pragma unroll
  for (int j = 0; j < 32; j += 8)
    tile[ty + j][tx] = t[((size_t)b * HW + (l0 + ty + j)) * C + (c0 + tx)];
  __syncthreads();
  #pragma unroll
  for (int j = 0; j < 32; j += 8)
    o[((size_t)b * C + (c0 + ty + j)) * HW + (l0 + tx)] = tile[tx][ty + j];
}

// ---------------------------------------------------------------------------
// Depthwise causal conv1d (k=4, left pad 3) + bias + SiLU.
// input = first di channels of xz (b, L, 2*di); output xc (b, L, di) fp32
// ---------------------------------------------------------------------------
__global__ void conv_silu_kernel(const float* __restrict__ xz, const float* __restrict__ cw,
                                 const float* __restrict__ cb, float* __restrict__ xc,
                                 int Bb, int L, int di) {
  size_t idx = (size_t)blockIdx.x * blockDim.x + threadIdx.x;
  size_t total = (size_t)Bb * L * di;
  if (idx >= total) return;
  const int ch = (int)(idx % di);
  const int l  = (int)((idx / di) % L);
  const int b  = (int)(idx / ((size_t)di * L));
  const int twodi = 2 * di;
  const size_t rowbase = (size_t)b * L * twodi;
  float acc = cb[ch];
  #pragma unroll
  for (int j = 0; j < 4; ++j) {
    int ll = l - 3 + j;
    if (ll >= 0) acc += cw[ch * 4 + j] * xz[rowbase + (size_t)ll * twodi + ch];
  }
  xc[idx] = silu_f(acc);
}

// ---------------------------------------------------------------------------
// Small projections: xdbl = xc @ wx^T, dt = softplus(xdbl[:R]@wdt^T+bdt),
// B = xdbl[R:R+2], C = xdbl[R+2:R+4]. Templated so xd[] stays in registers.
// ---------------------------------------------------------------------------
template <int DI, int R>
__global__ __launch_bounds__(128)
void smallproj_kernel(const float* __restrict__ xc, const float* __restrict__ wx,
                      const float* __restrict__ wdt, const float* __restrict__ bdt,
                      float* __restrict__ dt, float* __restrict__ Bc,
                      float* __restrict__ Cc, int P) {
  constexpr int R2N = R + 4;   // n = 2
  __shared__ float wxs[R2N * DI];
  __shared__ float wdts[DI * R];
  __shared__ float bdts[DI];
  for (int i = threadIdx.x; i < R2N * DI; i += blockDim.x) wxs[i] = wx[i];
  for (int i = threadIdx.x; i < DI * R;  i += blockDim.x) wdts[i] = wdt[i];
  for (int i = threadIdx.x; i < DI;      i += blockDim.x) bdts[i] = bdt[i];
  __syncthreads();

  const int pos = blockIdx.x * blockDim.x + threadIdx.x;
  if (pos >= P) return;

  float xd[R2N];
  #pragma unroll
  for (int j = 0; j < R2N; ++j) xd[j] = 0.0f;
  const float* row = xc + (size_t)pos * DI;
  for (int ch = 0; ch < DI; ++ch) {
    float v = row[ch];
    #pragma unroll
    for (int j = 0; j < R2N; ++j) xd[j] += v * wxs[j * DI + ch];
  }
  for (int i = 0; i < DI; ++i) {
    float a = bdts[i];
    #pragma unroll
    for (int j = 0; j < R; ++j) a += xd[j] * wdts[i * R + j];
    dt[(size_t)pos * DI + i] = softplus_f(a);
  }
  #pragma unroll
  for (int s = 0; s < 2; ++s) {
    Bc[(size_t)pos * 2 + s] = xd[R + s];
    Cc[(size_t)pos * 2 + s] = xd[R + 2 + s];
  }
}

// ---------------------------------------------------------------------------
// Chunked parallel scan: h_l = exp(dt*A)*h_{l-1} + dt*B*xc, n = 2 states.
// ---------------------------------------------------------------------------
#define NCHUNK 64
#define CSZ    64

__global__ __launch_bounds__(256)
void scan_chunk_kernel(const float* __restrict__ dt, const float* __restrict__ xc,
                       const float* __restrict__ Bc, const float* __restrict__ alog,
                       float* __restrict__ cA, float* __restrict__ cB,
                       int Bb, int L, int di) {
  int idx = blockIdx.x * blockDim.x + threadIdx.x;
  int total = Bb * NCHUNK * 2 * di;
  if (idx >= total) return;
  const int ch    = idx % di;
  const int s     = (idx / di) & 1;
  const int chunk = (idx / (2 * di)) % NCHUNK;
  const int b     = idx / (2 * di * NCHUNK);
  const float Av = -__expf(alog[ch * 2 + s]);
  float a = 1.0f, h = 0.0f;
  const size_t pbase = (size_t)b * L + (size_t)chunk * CSZ;
  for (int k = 0; k < CSZ; ++k) {
    const size_t p = pbase + k;
    const float dtv = dt[p * di + ch];
    const float xcv = xc[p * di + ch];
    const float bv  = Bc[p * 2 + s];
    const float e = __expf(dtv * Av);
    a *= e;
    h = e * h + dtv * bv * xcv;
  }
  cA[idx] = a;
  cB[idx] = h;
}

__global__ __launch_bounds__(256)
void scan_prefix_kernel(const float* __restrict__ cA, const float* __restrict__ cB,
                        float* __restrict__ pf, int Bb, int di) {
  int idx = blockIdx.x * blockDim.x + threadIdx.x;
  int total = Bb * 2 * di;
  if (idx >= total) return;
  const int ch = idx % di;
  const int s  = (idx / di) & 1;
  const int b  = idx / (2 * di);
  float h = 0.0f;
  for (int c = 0; c < NCHUNK; ++c) {
    const int q = (((b * NCHUNK + c) * 2) + s) * di + ch;
    pf[q] = h;
    h = cA[q] * h + cB[q];
  }
}

__global__ __launch_bounds__(256)
void scan_apply_kernel(const float* __restrict__ dt, const float* __restrict__ xc,
                       const float* __restrict__ Bc, const float* __restrict__ Cc,
                       const float* __restrict__ xz, const float* __restrict__ alog,
                       const float* __restrict__ dd, const float* __restrict__ pf,
                       unsigned short* __restrict__ y, int Bb, int L, int di) {
  int idx = blockIdx.x * blockDim.x + threadIdx.x;
  int total = Bb * NCHUNK * di;
  if (idx >= total) return;
  const int ch    = idx % di;
  const int chunk = (idx / di) % NCHUNK;
  const int b     = idx / (di * NCHUNK);
  const float A0 = -__expf(alog[ch * 2 + 0]);
  const float A1 = -__expf(alog[ch * 2 + 1]);
  const int qb = (b * NCHUNK + chunk) * 2;
  float h0 = pf[(size_t)(qb + 0) * di + ch];
  float h1 = pf[(size_t)(qb + 1) * di + ch];
  const float ddv = dd[ch];
  const int twodi = 2 * di;
  const size_t pbase = (size_t)b * L + (size_t)chunk * CSZ;
  for (int k = 0; k < CSZ; ++k) {
    const size_t p = pbase + k;
    const float dtv = dt[p * di + ch];
    const float xcv = xc[p * di + ch];
    const float b0 = Bc[p * 2 + 0], b1 = Bc[p * 2 + 1];
    const float c0 = Cc[p * 2 + 0], c1 = Cc[p * 2 + 1];
    h0 = __expf(dtv * A0) * h0 + dtv * b0 * xcv;
    h1 = __expf(dtv * A1) * h1 + dtv * b1 * xcv;
    float yv = h0 * c0 + h1 * c1 + ddv * xcv;
    const float zv = xz[p * twodi + di + ch];
    y[p * di + ch] = f2bf(yv * silu_f(zv));    // bf16 straight into next GEMM
  }
}

// ---------------------------------------------------------------------------
// LayerNorm over last dim D (blockDim == D, D <= 256); fp32 and/or bf16 out.
// ---------------------------------------------------------------------------
__global__ void layernorm_kernel(const float* __restrict__ in,
                                 float* __restrict__ outf,
                                 unsigned short* __restrict__ outb,
                                 const float* __restrict__ w,
                                 const float* __restrict__ bvec, int D) {
  __shared__ float s1[256];
  __shared__ float s2[256];
  const int row = blockIdx.x, t = threadIdx.x;
  const float v = in[(size_t)row * D + t];
  s1[t] = v;
  s2[t] = v * v;
  __syncthreads();
  for (int off = D >> 1; off > 0; off >>= 1) {
    if (t < off) { s1[t] += s1[t + off]; s2[t] += s2[t + off]; }
    __syncthreads();
  }
  const float mu  = s1[0] / (float)D;
  const float var = s2[0] / (float)D - mu * mu;
  const float inv = rsqrtf(var + 1e-5f);
  const float r = (v - mu) * inv * w[t] + bvec[t];
  if (outf) outf[(size_t)row * D + t] = r;
  if (outb) outb[(size_t)row * D + t] = f2bf(r);
}

// ---------------------------------------------------------------------------
// Host orchestration.
// ---------------------------------------------------------------------------
extern "C" void kernel_launch(void* const* d_in, const int* in_sizes, int n_in,
                              void* d_out, int out_size, void* d_ws, size_t ws_size,
                              hipStream_t stream) {
  (void)in_sizes; (void)n_in; (void)out_size; (void)ws_size;

  const float* x      = (const float*)d_in[0];
  const float* lin_w  = (const float*)d_in[1];
  const float* lin_b  = (const float*)d_in[2];
  const float* s1_win = (const float*)d_in[3];
  const float* s1_bin = (const float*)d_in[4];
  const float* s1_cw  = (const float*)d_in[5];
  const float* s1_cb  = (const float*)d_in[6];
  const float* s1_wx  = (const float*)d_in[7];
  const float* s1_wdt = (const float*)d_in[8];
  const float* s1_bdt = (const float*)d_in[9];
  const float* s1_alog= (const float*)d_in[10];
  const float* s1_dd  = (const float*)d_in[11];
  const float* s1_wout= (const float*)d_in[12];
  const float* s1_lnw = (const float*)d_in[13];
  const float* s1_lnb = (const float*)d_in[14];
  const float* s2_win = (const float*)d_in[15];
  const float* s2_bin = (const float*)d_in[16];
  const float* s2_cw  = (const float*)d_in[17];
  const float* s2_cb  = (const float*)d_in[18];
  const float* s2_wx  = (const float*)d_in[19];
  const float* s2_wdt = (const float*)d_in[20];
  const float* s2_bdt = (const float*)d_in[21];
  const float* s2_alog= (const float*)d_in[22];
  const float* s2_dd  = (const float*)d_in[23];
  const float* s2_wout= (const float*)d_in[24];
  const float* s2_lnw = (const float*)d_in[25];
  const float* s2_lnb = (const float*)d_in[26];

  const int B = 8, HW = 4096, C1 = 128, C2 = 256;
  const int M = B * HW;                       // 32768 token rows

  char* wsb = (char*)d_ws;
  auto F = [&](size_t off) { return (float*)(wsb + off); };
  auto U = [&](size_t off) { return (unsigned short*)(wsb + off); };

  // ---- persistent bf16 weights --------------------------------------------
  const size_t wb_win1  = 0;
  const size_t wb_wout1 = wb_win1  + (size_t)2 * C1 * C1 * 2;
  const size_t wb_lin   = wb_wout1 + (size_t)C1 * C1 * 2;
  const size_t wb_win2  = wb_lin   + (size_t)C2 * C1 * 2;
  const size_t wb_wout2 = wb_win2  + (size_t)2 * C2 * C2 * 2;
  size_t S0             = wb_wout2 + (size_t)C2 * C2 * 2;
  S0 = (S0 + 255) & ~(size_t)255;

  // ---- arena (byte offsets), lifetime-based aliasing ----------------------
  const size_t o_t1bf = S0;                                   // (M,128) bf16
  const size_t o_xz1  = o_t1bf + (size_t)M * C1 * 2;          // (M,256) f32
  const size_t o_xc1  = o_xz1  + (size_t)M * 2 * C1 * 4;      // (M,128) f32
  const size_t o_dt1  = o_xc1  + (size_t)M * C1 * 4;          // (M,128) f32
  const size_t o_B1   = o_dt1  + (size_t)M * C1 * 4;
  const size_t o_C1v  = o_B1   + (size_t)M * 2 * 4;
  const size_t o_cA1  = o_C1v  + (size_t)M * 2 * 4;
  const size_t o_cB1  = o_cA1  + (size_t)B * NCHUNK * 2 * C1 * 4;
  const size_t o_pf1  = o_cB1  + (size_t)B * NCHUNK * 2 * C1 * 4;
  const size_t o_y1bf = o_pf1  + (size_t)B * NCHUNK * 2 * C1 * 4; // (M,128) bf16
  const size_t o_o1   = o_y1bf + (size_t)M * C1 * 2;          // (M,128) f32
  const size_t o_o1bf = o_o1   + (size_t)M * C1 * 4;          // (M,128) bf16
  // stage 2 over dead stage-1 regions
  const size_t o_a2bf = S0;                                   // (M,256) bf16
  const size_t o_xz2  = S0 + (size_t)M * C2 * 2;              // (M,512) f32
  const size_t o_xc2  = o_o1;                                 // (M,256) f32
  const size_t o_dt2  = o_xc2 + (size_t)M * C2 * 4;           // (M,256) f32
  const size_t o_B2   = o_dt2 + (size_t)M * C2 * 4;
  const size_t o_C2v  = o_B2  + (size_t)M * 2 * 4;
  const size_t o_cA2  = o_C2v + (size_t)M * 2 * 4;
  const size_t o_cB2  = o_cA2 + (size_t)B * NCHUNK * 2 * C2 * 4;
  const size_t o_pf2  = o_cB2 + (size_t)B * NCHUNK * 2 * C2 * 4;
  const size_t o_y2bf = o_pf2 + (size_t)B * NCHUNK * 2 * C2 * 4;  // (M,256) bf16
  const size_t o_o2   = S0;                                   // (M,256) f32
  // peak ~ S0 + 173.5 MB

  // ---- weight conversion ---------------------------------------------------
  auto cvt = [&](const float* src, size_t off, int n) {
    f32_to_bf16_kernel<<<(n + 255) / 256, 256, 0, stream>>>(src, U(off), n);
  };
  cvt(s1_win,  wb_win1,  2 * C1 * C1);
  cvt(s1_wout, wb_wout1, C1 * C1);
  cvt(lin_w,   wb_lin,   C2 * C1);
  cvt(s2_win,  wb_win2,  2 * C2 * C2);
  cvt(s2_wout, wb_wout2, C2 * C2);

  const dim3 tb32x8(32, 8);

  // ---- stage 1 -------------------------------------------------------------
  transpose_in_kernel<<<dim3(HW / 32, C1 / 32, B), tb32x8, 0, stream>>>(
      x, U(o_t1bf), C1, HW);

  gemm_bf16_wmma<<<dim3(M / 64, (2 * C1) / 64), 128, 0, stream>>>(
      U(o_t1bf), U(wb_win1), s1_bin, F(o_xz1), nullptr, M, 2 * C1, C1, 0);

  {
    size_t tot = (size_t)M * C1;
    conv_silu_kernel<<<(unsigned)((tot + 255) / 256), 256, 0, stream>>>(
        F(o_xz1), s1_cw, s1_cb, F(o_xc1), B, HW, C1);
  }

  smallproj_kernel<128, 8><<<(M + 127) / 128, 128, 0, stream>>>(
      F(o_xc1), s1_wx, s1_wdt, s1_bdt, F(o_dt1), F(o_B1), F(o_C1v), M);

  {
    int tot = B * NCHUNK * 2 * C1;
    scan_chunk_kernel<<<(tot + 255) / 256, 256, 0, stream>>>(
        F(o_dt1), F(o_xc1), F(o_B1), s1_alog, F(o_cA1), F(o_cB1), B, HW, C1);
  }
  {
    int tot = B * 2 * C1;
    scan_prefix_kernel<<<(tot + 255) / 256, 256, 0, stream>>>(
        F(o_cA1), F(o_cB1), F(o_pf1), B, C1);
  }
  {
    int tot = B * NCHUNK * C1;
    scan_apply_kernel<<<(tot + 255) / 256, 256, 0, stream>>>(
        F(o_dt1), F(o_xc1), F(o_B1), F(o_C1v), F(o_xz1), s1_alog, s1_dd,
        F(o_pf1), U(o_y1bf), B, HW, C1);
  }

  gemm_bf16_wmma<<<dim3(M / 64, C1 / 64), 128, 0, stream>>>(
      U(o_y1bf), U(wb_wout1), nullptr, F(o_o1), nullptr, M, C1, C1, 0);

  layernorm_kernel<<<M, C1, 0, stream>>>(F(o_o1), nullptr, U(o_o1bf),
                                         s1_lnw, s1_lnb, C1);

  // ---- inter-stage linear + SiLU (bf16 out straight into next GEMM) -------
  gemm_bf16_wmma<<<dim3(M / 64, C2 / 64), 128, 0, stream>>>(
      U(o_o1bf), U(wb_lin), lin_b, nullptr, U(o_a2bf), M, C2, C1, 1);

  // ---- stage 2 -------------------------------------------------------------
  gemm_bf16_wmma<<<dim3(M / 64, (2 * C2) / 64), 128, 0, stream>>>(
      U(o_a2bf), U(wb_win2), s2_bin, F(o_xz2), nullptr, M, 2 * C2, C2, 0);

  {
    size_t tot = (size_t)M * C2;
    conv_silu_kernel<<<(unsigned)((tot + 255) / 256), 256, 0, stream>>>(
        F(o_xz2), s2_cw, s2_cb, F(o_xc2), B, HW, C2);
  }

  smallproj_kernel<256, 16><<<(M + 127) / 128, 128, 0, stream>>>(
      F(o_xc2), s2_wx, s2_wdt, s2_bdt, F(o_dt2), F(o_B2), F(o_C2v), M);

  {
    int tot = B * NCHUNK * 2 * C2;
    scan_chunk_kernel<<<(tot + 255) / 256, 256, 0, stream>>>(
        F(o_dt2), F(o_xc2), F(o_B2), s2_alog, F(o_cA2), F(o_cB2), B, HW, C2);
  }
  {
    int tot = B * 2 * C2;
    scan_prefix_kernel<<<(tot + 255) / 256, 256, 0, stream>>>(
        F(o_cA2), F(o_cB2), F(o_pf2), B, C2);
  }
  {
    int tot = B * NCHUNK * C2;
    scan_apply_kernel<<<(tot + 255) / 256, 256, 0, stream>>>(
        F(o_dt2), F(o_xc2), F(o_B2), F(o_C2v), F(o_xz2), s2_alog, s2_dd,
        F(o_pf2), U(o_y2bf), B, HW, C2);
  }

  gemm_bf16_wmma<<<dim3(M / 64, C2 / 64), 128, 0, stream>>>(
      U(o_y2bf), U(wb_wout2), nullptr, F(o_o2), nullptr, M, C2, C2, 0);

  layernorm_kernel<<<M, C2, 0, stream>>>(F(o_o2), F(o_o2), nullptr,
                                         s2_lnw, s2_lnb, C2);

  transpose_out_kernel<<<dim3(HW / 32, C2 / 32, B), tb32x8, 0, stream>>>(
      F(o_o2), (float*)d_out, C2, HW);
}
